// MultiHeadedSelfAttention_24352464568540
// MI455X (gfx1250) — compile-verified, matrix operand order checked by reading, not run on hardware
//
#include <hip/hip_runtime.h>
#include <cstdint>

// ---------------------------------------------------------------------------
// MHA for MI455X (gfx1250): bf16 WMMA (16x16x32, f32 accum), TDM tensor load,
// double-buffered async-to-LDS, all LDS fragment traffic as b128,
// hardware f32->bf16 conversion.
// Shapes: B=4, S=2048, D=1024, H=16, W=64.
// ---------------------------------------------------------------------------

typedef __attribute__((ext_vector_type(16))) __bf16 v16bf;
typedef __attribute__((ext_vector_type(8)))  __bf16 v8bf;
typedef __attribute__((ext_vector_type(2)))  __bf16 v2bf;
typedef __attribute__((ext_vector_type(8)))  float  v8f;
typedef __attribute__((ext_vector_type(4)))  unsigned int v4u;
typedef __attribute__((ext_vector_type(8)))  int    v8i;
typedef __attribute__((ext_vector_type(4)))  int    v4i;

#define B_  4
#define S_  2048
#define D_  1024
#define H_  16
#define W_  64
#define M_  (B_ * S_)   // 8192 rows of x

// Hardware f32 -> bf16 (backend emits v_cvt_*bf16_f32, RNE).
__device__ __forceinline__ unsigned short f2bf(float f) {
  return __builtin_bit_cast(unsigned short, (__bf16)f);
}
// Packed pair conversion: lets the backend match v_cvt_pk_bf16_f32.
__device__ __forceinline__ unsigned int f2bf2(float lo, float hi) {
  v2bf p = { (__bf16)lo, (__bf16)hi };
  return __builtin_bit_cast(unsigned int, p);
}
__device__ __forceinline__ v8f wmma_bf16(v16bf a, v16bf b, v8f c) {
  return __builtin_amdgcn_wmma_f32_16x16x32_bf16(
      false, a, false, b, (short)0, c, false, false);
}
__device__ __forceinline__ v8f vzero8() {
  v8f z = {0.f, 0.f, 0.f, 0.f, 0.f, 0.f, 0.f, 0.f};
  return z;
}
// Two aligned 16B LDS reads -> one 16-element bf16 fragment (ds_load_b128 x2).
__device__ __forceinline__ v16bf frag_ld(const unsigned short* p_lo,
                                         const unsigned short* p_hi) {
  v8bf lo = *reinterpret_cast<const v8bf*>(p_lo);
  v8bf hi = *reinterpret_cast<const v8bf*>(p_hi);
  return __builtin_shufflevector(lo, hi, 0, 1, 2, 3, 4, 5, 6, 7,
                                         8, 9, 10, 11, 12, 13, 14, 15);
}

// ---------------------------------------------------------------------------
// Kernel 1: Q/K/V projection GEMM.  C[8192x1024] = x @ W + b.
// Q,K stored bf16 head-major [B*H][S][W]; V stored TRANSPOSED [B*H][W][S] so
// the attention P@V B-fragment is a contiguous row read.
// Block tile 128x64, 8 waves of 32x32 (2x2 WMMA tiles), K-step 32.
// ---------------------------------------------------------------------------
__global__ __launch_bounds__(256) void qkv_gemm_kernel(
    const float* __restrict__ x,
    const float* __restrict__ Wq, const float* __restrict__ bq,
    const float* __restrict__ Wk, const float* __restrict__ bk,
    const float* __restrict__ Wv, const float* __restrict__ bv,
    unsigned short* __restrict__ qws,
    unsigned short* __restrict__ kws,
    unsigned short* __restrict__ vws)
{
  const float* Wm  = (blockIdx.z == 0) ? Wq : (blockIdx.z == 1) ? Wk : Wv;
  const float* bia = (blockIdx.z == 0) ? bq : (blockIdx.z == 1) ? bk : bv;
  unsigned short* out = (blockIdx.z == 0) ? qws : (blockIdx.z == 1) ? kws : vws;

  __shared__ unsigned short Xs[128][40];   // 128 x 32 bf16, 80B rows (16B mult)
  __shared__ unsigned short Wt[64][40];    // transposed W tile: [n][k]

  const int t    = threadIdx.x;
  const int wid  = t >> 5;
  const int lane = t & 31;
  const int half = lane >> 4;
  const int l15  = lane & 15;
  const int kb2  = half * 8;

  const int m_blk = blockIdx.y * 128;
  const int n_blk = blockIdx.x * 64;
  const int wm = (wid >> 1) * 32;
  const int wn = (wid & 1) * 32;

  v8f acc[2][2];
  #pragma unroll
  for (int i = 0; i < 2; ++i)
    #pragma unroll
    for (int j = 0; j < 2; ++j) acc[i][j] = vzero8();

  const int xr = t >> 1;            // 0..127
  const int xc = (t & 1) * 16;      // 0 or 16
  const int wr = t >> 3;            // 0..31  (k row of W)
  const int wc = (t & 7) * 8;       // 0..56  (n col of W)

  for (int kb = 0; kb < D_; kb += 32) {
    // stage x tile: fp32 -> packed bf16 pairs -> ds_store_b64
    const float* xp = x + (size_t)(m_blk + xr) * D_ + kb + xc;
    #pragma unroll
    for (int j = 0; j < 16; j += 4) {
      float4 f = *(const float4*)(xp + j);
      uint2 p;
      p.x = f2bf2(f.x, f.y);
      p.y = f2bf2(f.z, f.w);
      *(uint2*)&Xs[xr][xc + j] = p;
    }
    // stage W tile transposed: Wt[n][k] = W[k][n]
    const float* wp = Wm + (size_t)(kb + wr) * D_ + n_blk + wc;
    #pragma unroll
    for (int j = 0; j < 8; j += 4) {
      float4 f = *(const float4*)(wp + j);
      Wt[wc + j + 0][wr] = f2bf(f.x);
      Wt[wc + j + 1][wr] = f2bf(f.y);
      Wt[wc + j + 2][wr] = f2bf(f.z);
      Wt[wc + j + 3][wr] = f2bf(f.w);
    }
    // prefetch next K-step tiles into cache (global_prefetch_b8)
    if (kb + 32 < D_) {
      __builtin_prefetch(xp + 32, 0, 0);
      __builtin_prefetch(wp + 32 * D_, 0, 0);
    }
    __syncthreads();

    v16bf afr[2], bfr[2];
    #pragma unroll
    for (int tm = 0; tm < 2; ++tm) {
      const int row = wm + tm * 16 + l15;
      afr[tm] = frag_ld(&Xs[row][kb2], &Xs[row][16 + kb2]);
    }
    #pragma unroll
    for (int tn = 0; tn < 2; ++tn) {
      const int col = wn + tn * 16 + l15;
      bfr[tn] = frag_ld(&Wt[col][kb2], &Wt[col][16 + kb2]);
    }
    #pragma unroll
    for (int tm = 0; tm < 2; ++tm)
      #pragma unroll
      for (int tn = 0; tn < 2; ++tn)
        acc[tm][tn] = wmma_bf16(afr[tm], bfr[tn], acc[tm][tn]);
    __syncthreads();
  }

  // epilogue: C row = half*8+r, col = l15; bias; remap (V transposed)
  #pragma unroll
  for (int tm = 0; tm < 2; ++tm)
    #pragma unroll
    for (int tn = 0; tn < 2; ++tn)
      #pragma unroll
      for (int r = 0; r < 8; ++r) {
        const int gm = m_blk + wm + tm * 16 + half * 8 + r;  // b*S + s
        const int gn = n_blk + wn + tn * 16 + l15;           // h*W + w
        const float v = acc[tm][tn][r] + bia[gn];
        const int b = gm >> 11, s = gm & (S_ - 1);
        const int h = gn >> 6,  w = gn & 63;
        size_t oi;
        if (blockIdx.z == 2)
          oi = (((size_t)(b * H_ + h)) * W_ + w) * S_ + s;   // V: [bh][w][s]
        else
          oi = (((size_t)(b * H_ + h)) * S_ + s) * W_ + w;   // Q,K: [bh][s][w]
        out[oi] = f2bf(v);
      }
}

// ---------------------------------------------------------------------------
// Kernel 2: flash-style attention.  Grid: (S/64, B*H).  128 threads = 4 waves,
// each wave owns 16 query rows.  Q tile loaded by TDM (tensor_load_to_lds,
// TENSORcnt); K/V 64-key blocks double-buffered via async-to-LDS (ASYNCcnt).
// ---------------------------------------------------------------------------
__global__ __launch_bounds__(128) void attention_kernel(
    const unsigned short* __restrict__ qws,
    const unsigned short* __restrict__ kws,
    const unsigned short* __restrict__ vws,
    const int* __restrict__ mask,
    float* __restrict__ out)
{
  __shared__ unsigned short Qs[64][72];        // 144B rows (16B multiple)
  __shared__ unsigned short Ks[2][64][72];     // [buf][key][w]
  __shared__ unsigned short Vs[2][64][72];     // [buf][w][key]  (V transposed)
  __shared__ unsigned short Ps[4][16][72];     // per-wave P tile

  const int t    = threadIdx.x;
  const int wid  = t >> 5;
  const int lane = t & 31;
  const int half = lane >> 4;
  const int l15  = lane & 15;
  const int kb2  = half * 8;

  const int bh   = blockIdx.y;                 // b*H + h
  const int bidx = bh >> 4;
  const int h    = bh & 15;
  const int q0   = blockIdx.x * 64;
  const size_t base = (size_t)bh * S_ * W_;    // same footprint for Q/K and Vt

  // issue K (row-major) and V (transposed) 64x64 tiles: 8 async b128 per wave
  auto issue_kv = [&](int buf, int kblk) {
    #pragma unroll
    for (int i = 0; i < 4; ++i) {
      const int lin = t + i * 128;             // 0..511 16B chunks
      const int row = lin >> 3;
      const int c8  = (lin & 7) * 8;
      unsigned long long gk =
          (unsigned long long)(kws + base + (size_t)(kblk + row) * W_ + c8);
      unsigned int lk = (unsigned int)(size_t)&Ks[buf][row][c8];
      asm volatile("global_load_async_to_lds_b128 %0, %1, off"
                   :: "v"(lk), "v"(gk) : "memory");
      unsigned long long gv =
          (unsigned long long)(vws + base + (size_t)row * S_ + kblk + c8);
      unsigned int lv = (unsigned int)(size_t)&Vs[buf][row][c8];
      asm volatile("global_load_async_to_lds_b128 %0, %1, off"
                   :: "v"(lv), "v"(gv) : "memory");
    }
  };

  issue_kv(0, 0);

  // --- Q tile via Tensor Data Mover: 2D tile 64x64 x 2B, pad 16B per 128B ---
  if (wid == 0) {
    const unsigned long long ga =
        (unsigned long long)(qws + base + (size_t)q0 * W_);
    v4u g0 = { 1u,                                   // count=1, user desc
               (unsigned int)(size_t)&Qs[0][0],      // lds_addr
               (unsigned int)(ga & 0xFFFFFFFFull),   // global_addr[31:0]
               (unsigned int)((ga >> 32) & 0x1FFFFFFull) | (2u << 30) }; // +type=2
    // group1: data_size=2B(1), pad_enable, pad_interval=32dw(4), pad_amount=4dw(3)
    const unsigned int w0 = (1u << 16) | (1u << 20) | (4u << 22) | (3u << 25);
    v8i g1 = { (int)w0,
               (int)(64u << 16),          // tensor_dim0 = 64 (low 16 in [31:16])
               (int)(2048u << 16),        // tensor_dim0 hi=0 | tensor_dim1=2048
               (int)(64u << 16),          // tensor_dim1 hi=0 | tile_dim0=64
               (int)64,                   // tile_dim1=64 | tile_dim2=0
               (int)64,                   // tensor_dim0_stride = 64
               0, 0 };                    // stride0 hi | tensor_dim1_stride = 0
    v4i z4 = {0, 0, 0, 0};
#if defined(__clang_major__) && (__clang_major__ >= 23)
    v8i z8 = {0, 0, 0, 0, 0, 0, 0, 0};
    __builtin_amdgcn_tensor_load_to_lds(g0, g1, z4, z4, z8, 0);
#else
    __builtin_amdgcn_tensor_load_to_lds(g0, g1, z4, z4, 0);
#endif
    __builtin_amdgcn_s_wait_tensorcnt(0);
  }

  float Mrun[8], Lrun[8];
  v8f o[4];
  #pragma unroll
  for (int r = 0; r < 8; ++r) { Mrun[r] = -1e30f; Lrun[r] = 0.f; }
  #pragma unroll
  for (int tw = 0; tw < 4; ++tw) o[tw] = vzero8();

  for (int kb = 0; kb < S_; kb += 64) {
    const int buf = (kb >> 6) & 1;
    // prefetch next block into the other buffer, wait only for current 8
    if (kb + 64 < S_) {
      issue_kv(buf ^ 1, kb + 64);
      asm volatile("s_wait_asynccnt 0x8" ::: "memory");
    } else {
      asm volatile("s_wait_asynccnt 0x0" ::: "memory");
    }
    __syncthreads();   // current tiles (and, first iter, TDM Q) visible to all

    // key-side additive mask penalty (per lane column)
    float pen[4];
    #pragma unroll
    for (int tn = 0; tn < 4; ++tn) {
      const int kpos = kb + tn * 16 + l15;
      pen[tn] = -10000.0f * (1.0f - (float)mask[bidx * S_ + kpos]);
    }

    // --- scores: Q(16x64) @ K^T -> 4 tiles of 16x16, K-steps of 32 ---
    v8f sacc[4];
    #pragma unroll
    for (int tn = 0; tn < 4; ++tn) sacc[tn] = vzero8();
    #pragma unroll
    for (int ks = 0; ks < 2; ++ks) {
      const int qrow = wid * 16 + l15;
      v16bf aq = frag_ld(&Qs[qrow][ks * 32 + kb2],
                         &Qs[qrow][ks * 32 + 16 + kb2]);
      #pragma unroll
      for (int tn = 0; tn < 4; ++tn) {
        const int col = tn * 16 + l15;          // key index in block
        v16bf bk_ = frag_ld(&Ks[buf][col][ks * 32 + kb2],
                            &Ks[buf][col][ks * 32 + 16 + kb2]);
        sacc[tn] = wmma_bf16(aq, bk_, sacc[tn]);
      }
    }

    // --- online softmax per query row (16-lane half-wave reductions) ---
    #pragma unroll
    for (int r = 0; r < 8; ++r) {
      float sc[4];
      float mx = -1e30f;
      #pragma unroll
      for (int tn = 0; tn < 4; ++tn) {
        sc[tn] = sacc[tn][r] * 0.125f + pen[tn];    // 1/sqrt(64)
        mx = fmaxf(mx, sc[tn]);
      }
      #pragma unroll
      for (int off = 1; off < 16; off <<= 1)
        mx = fmaxf(mx, __shfl_xor(mx, off, 32));
      const float Mnew  = fmaxf(Mrun[r], mx);
      const float scale = __expf(Mrun[r] - Mnew);
      float sum = 0.f;
      #pragma unroll
      for (int tn = 0; tn < 4; ++tn) {
        const float e = __expf(sc[tn] - Mnew);
        sum += e;
        Ps[wid][half * 8 + r][tn * 16 + l15] = f2bf(e);
      }
      #pragma unroll
      for (int off = 1; off < 16; off <<= 1)
        sum += __shfl_xor(sum, off, 32);
      Lrun[r] = Lrun[r] * scale + sum;
      Mrun[r] = Mnew;
      #pragma unroll
      for (int tw = 0; tw < 4; ++tw) o[tw][r] *= scale;
    }
    asm volatile("s_wait_dscnt 0" ::: "memory");  // P visible (same-wave DS order)

    // --- O += P(16x64) @ V(64x64), V tile stored transposed [w][k] ---
    #pragma unroll
    for (int ks = 0; ks < 2; ++ks) {
      v16bf ap = frag_ld(&Ps[wid][l15][ks * 32 + kb2],
                         &Ps[wid][l15][ks * 32 + 16 + kb2]);
      #pragma unroll
      for (int tw = 0; tw < 4; ++tw) {
        const int col = tw * 16 + l15;          // w index
        v16bf bv_ = frag_ld(&Vs[buf][col][ks * 32 + kb2],
                            &Vs[buf][col][ks * 32 + 16 + kb2]);
        o[tw] = wmma_bf16(ap, bv_, o[tw]);
      }
    }
    __syncthreads();   // all waves done with buf before it is re-issued
  }

  // --- final: out[b][s][h*64+w] = O / Lsum ---
  #pragma unroll
  for (int r = 0; r < 8; ++r) {
    const float inv = 1.0f / Lrun[r];
    const int q = q0 + wid * 16 + half * 8 + r;
    #pragma unroll
    for (int tw = 0; tw < 4; ++tw) {
      const int w = tw * 16 + l15;
      out[((size_t)bidx * S_ + q) * D_ + h * W_ + w] = o[tw][r] * inv;
    }
  }
}

// ---------------------------------------------------------------------------
extern "C" void kernel_launch(void* const* d_in, const int* in_sizes, int n_in,
                              void* d_out, int out_size, void* d_ws, size_t ws_size,
                              hipStream_t stream) {
  const float* x  = (const float*)d_in[0];
  const int*   mk = (const int*)  d_in[1];
  const float* Wq = (const float*)d_in[2];
  const float* bq = (const float*)d_in[3];
  const float* Wk = (const float*)d_in[4];
  const float* bk = (const float*)d_in[5];
  const float* Wv = (const float*)d_in[6];
  const float* bv = (const float*)d_in[7];
  float* out = (float*)d_out;

  // bf16 workspace: Q,K head-major [B*H][S][W]; V transposed [B*H][W][S]
  unsigned short* qws = (unsigned short*)d_ws;
  unsigned short* kws = qws + (size_t)M_ * D_;
  unsigned short* vws = kws + (size_t)M_ * D_;

  dim3 g1(D_ / 64, M_ / 128, 3);   // (16, 64, 3)
  qkv_gemm_kernel<<<g1, 256, 0, stream>>>(x, Wq, bq, Wk, bk, Wv, bv,
                                          qws, kws, vws);

  dim3 g2(S_ / 64, B_ * H_);       // (32, 64)
  attention_kernel<<<g2, 128, 0, stream>>>(qws, kws, vws, mk, out);
}
